// OfflineReconstructorJetLatentSet2SetK_38551626448916
// MI455X (gfx1250) — matrix operand order repaired and checked by reading, not compiled, not run on hardware
//
#include <hip/hip_runtime.h>
#include <math.h>

// ---------------------------------------------------------------------------
// OfflineReconstructorJetLatentSet2Set — MI455X (gfx1250, wave32, WMMA) port.
//
// Shapes: B=64, L=192, D=256, NH=8(dh=32), NL=8, FF=1024, IN=7
// Heavy GEMMs and attention matmuls run on v_wmma_f32_16x16x32_f16 with f32
// accumulation. GEMM tiles are staged global->LDS with GLOBAL_LOAD_ASYNC_TO_LDS
// (ASYNCcnt-pipelined, double-buffered) when the toolchain exposes the
// builtins; otherwise a synchronous b128 copy path is used.
// ---------------------------------------------------------------------------

#define BB   64
#define LL   192
#define DD   256
#define NH8  8
#define FF_  1024
#define IN7  7
#define NTOK (BB * LL)
#define EPSF 1e-8f

typedef __attribute__((ext_vector_type(16))) _Float16 v16h;
typedef __attribute__((ext_vector_type(2)))  __fp16   f16x2;
typedef __attribute__((ext_vector_type(8)))  float    v8f;
typedef __attribute__((ext_vector_type(4)))  float    v4f;
typedef int v4i_gcc __attribute__((vector_size(4 * sizeof(int))));

#if defined(__has_builtin)
#if __has_builtin(__builtin_amdgcn_global_load_async_to_lds_b128) && \
    __has_builtin(__builtin_amdgcn_global_load_async_to_lds_b32)  && \
    __has_builtin(__builtin_amdgcn_s_wait_asynccnt)
#define HAVE_ASYNC_LDS 1
#endif
#endif

#ifdef HAVE_ASYNC_LDS
// Param types per hipcc diagnostics: (AS1 typed data ptr, AS3 typed data ptr,
// imm offset, imm cpol). b128 uses int4 (vector_size 16), b32 uses int.
__device__ __forceinline__ void async_b128(const float* g, float* l) {
  __builtin_amdgcn_global_load_async_to_lds_b128(
      (__attribute__((address_space(1))) v4i_gcc*)g,
      (__attribute__((address_space(3))) v4i_gcc*)l, 0, 0);
}
__device__ __forceinline__ void async_b32(const float* g, float* l) {
  __builtin_amdgcn_global_load_async_to_lds_b32(
      (__attribute__((address_space(1))) int*)g,
      (__attribute__((address_space(3))) int*)l, 0, 0);
}
#endif

// ---------------- device helpers ----------------
__device__ __forceinline__ float geluf(float x) {
  return 0.5f * x * (1.0f + erff(x * 0.70710678118654752f));
}
__device__ __forceinline__ float clampf(float x, float lo, float hi) {
  return fminf(fmaxf(x, lo), hi);
}
__device__ __forceinline__ float wrap_phi(float p) {
  const float PI  = 3.14159265358979323846f;
  const float TPI = 6.28318530717958647692f;
  float t = p + PI;
  float q = t - TPI * floorf(t / TPI);
  return q - PI;
}
__device__ __forceinline__ float wave_sum(float v) {
  for (int o = 16; o; o >>= 1) v += __shfl_xor(v, o, 32);
  return v;
}
__device__ __forceinline__ float wave_max(float v) {
  for (int o = 16; o; o >>= 1) v = fmaxf(v, __shfl_xor(v, o, 32));
  return v;
}
__device__ __forceinline__ float apply_act(float v, int act) {
  if (act == 1) return geluf(v);
  if (act == 2) return tanhf(v);
  return v;
}

// Pack 16 f32 (four v4f chunks) into a v16h fragment via v_cvt_pk_rtz_f16_f32.
// cvt_pkrtz returns __fp16x2; reinterpret into the _Float16x16 WMMA fragment.
__device__ __forceinline__ v16h frag_pack(v4f f0, v4f f1, v4f f2, v4f f3) {
  union { v16h v; f16x2 p[8]; } u;
  u.p[0] = __builtin_amdgcn_cvt_pkrtz(f0[0], f0[1]);
  u.p[1] = __builtin_amdgcn_cvt_pkrtz(f0[2], f0[3]);
  u.p[2] = __builtin_amdgcn_cvt_pkrtz(f1[0], f1[1]);
  u.p[3] = __builtin_amdgcn_cvt_pkrtz(f1[2], f1[3]);
  u.p[4] = __builtin_amdgcn_cvt_pkrtz(f2[0], f2[1]);
  u.p[5] = __builtin_amdgcn_cvt_pkrtz(f2[2], f2[3]);
  u.p[6] = __builtin_amdgcn_cvt_pkrtz(f3[0], f3[1]);
  u.p[7] = __builtin_amdgcn_cvt_pkrtz(f3[2], f3[3]);
  return u.v;
}
// A-fragment from one 32-float K-row: elems 0..15 <- K=c0..c0+15
// (c0 = half?16:0). One 16B-aligned contiguous run -> b128 loads.
__device__ __forceinline__ v16h frag_a16(const float* row, int half) {
  const v4f* q0 = (const v4f*)(row + (half ? 16 : 0));
  return frag_pack(q0[0], q0[1], q0[2], q0[3]);
}
// B-fragment: 16 contiguous K values (lane = output column).
__device__ __forceinline__ v16h frag_b16(const float* p) {
  const v4f* q = (const v4f*)p;
  return frag_pack(q[0], q[1], q[2], q[3]);
}

// ---------------------------------------------------------------------------
// WMMA GEMM:  C[M,N] = act(A[M,K] * W[K,N] + bias[N])
// block = 128 threads (4 waves). Block tile 64(M) x 32(N); wave tile 16x32
// (one A fragment feeds two WMMAs). K step = 32. Tiles staged to LDS as f32
// (async copies can't convert), double-buffered, converted to f16 during
// fragment build. W is staged TRANSPOSED so B fragments are contiguous.
// Requires M%64==0, N%32==0, K%32==0.
// ---------------------------------------------------------------------------
__global__ __launch_bounds__(128) void wmma_gemm_kernel(
    const float* __restrict__ A, const float* __restrict__ W,
    const float* __restrict__ bias, float* __restrict__ C,
    int M, int N, int K, int act)
{
  __shared__ __align__(16) float sA[2][64 * 32];   // [m][k] row-major
  __shared__ __align__(16) float sW[2][32 * 32];   // transposed: [n][k]
  const int tid  = threadIdx.x;
  const int wave = tid >> 5;
  const int lane = tid & 31;
  const int half = lane >> 4;
  const int l15  = lane & 15;
  const int m0 = blockIdx.y * 64;
  const int n0 = blockIdx.x * 32;

  // Stage one K-tile: A = 4 b128 ops/thread, W = 8 b32 ops/thread (scatter
  // transpose — each lane supplies its own LDS address). 12 async ops/wave.
  auto stage = [&](int buf, int k0) {
    #pragma unroll
    for (int it = 0; it < 4; ++it) {
      int c = tid + it * 128;               // chunk 0..511 (16B each)
      int r = c >> 3, c4 = (c & 7) << 2;
      const float* g = A + (size_t)(m0 + r) * K + k0 + c4;
      float* l = &sA[buf][r * 32 + c4];
#ifdef HAVE_ASYNC_LDS
      async_b128(g, l);
#else
      *(v4f*)l = *(const v4f*)g;
#endif
    }
    #pragma unroll
    for (int it = 0; it < 8; ++it) {
      int e = tid + it * 128;               // element 0..1023
      int r = e >> 5, cn = e & 31;          // r = K row, cn = N col
      const float* g = W + (size_t)(k0 + r) * N + n0 + cn;
      float* l = &sW[buf][cn * 32 + r];
#ifdef HAVE_ASYNC_LDS
      async_b32(g, l);
#else
      *l = *g;
#endif
    }
  };

  v8f acc0 = {}, acc1 = {};
  int buf = 0;
  stage(0, 0);
  for (int k0 = 0; k0 < K; k0 += 32) {
    const bool haveNext = (k0 + 32 < K);
    if (haveNext) {
      stage(buf ^ 1, k0 + 32);              // prefetch next tile (other buffer)
#ifdef HAVE_ASYNC_LDS
      __builtin_amdgcn_s_wait_asynccnt(12); // current tile done; next in flight
#endif
    } else {
#ifdef HAVE_ASYNC_LDS
      __builtin_amdgcn_s_wait_asynccnt(0);
#endif
    }
    __syncthreads();

    const float* arow = &sA[buf][((wave << 4) + l15) * 32];
    v16h af  = frag_a16(arow, half);
    v16h bf0 = frag_b16(&sW[buf][ l15        * 32 + (half ? 16 : 0)]);
    v16h bf1 = frag_b16(&sW[buf][(l15 + 16)  * 32 + (half ? 16 : 0)]);
    acc0 = __builtin_amdgcn_wmma_f32_16x16x32_f16(
        false, af, false, bf0, (short)0, acc0, false, false);
    acc1 = __builtin_amdgcn_wmma_f32_16x16x32_f16(
        false, af, false, bf1, (short)0, acc1, false, false);
    __syncthreads();                        // protect buf before re-stage
    buf ^= 1;
  }

  // C layout: VGPR r -> row r (lanes 0-15) / r+8 (lanes 16-31); col = l15
  const int rbase = m0 + (wave << 4) + (half ? 8 : 0);
  const int col0  = n0 + l15;
  const int col1  = n0 + 16 + l15;
  const float bv0 = bias ? bias[col0] : 0.0f;
  const float bv1 = bias ? bias[col1] : 0.0f;
  #pragma unroll
  for (int r = 0; r < 8; ++r) {
    C[(size_t)(rbase + r) * N + col0] = apply_act(acc0[r] + bv0, act);
    C[(size_t)(rbase + r) * N + col1] = apply_act(acc1[r] + bv1, act);
  }
}

// ---------------------------------------------------------------------------
// Attention scores: S[b,h,qi,kj] = scale * Q[b,qi,h,:] . K[b,kj,h,:]  (+bias)
// Q,K token-major (B*L, nh*dh). One wave per 16x16 score tile. Q/K rows are
// contiguous in the fragment K-order -> vector loads + cvt_pkrtz.
// ---------------------------------------------------------------------------
__global__ __launch_bounds__(32) void attn_scores_kernel(
    const float* __restrict__ Q, const float* __restrict__ Km,
    const float* __restrict__ bias, float* __restrict__ S,
    int Lq, int nh, int dh, float scale)
{
  const int bh = blockIdx.z;
  const int b = bh / nh, h = bh - b * nh;
  const int qi = blockIdx.y * 16, kj = blockIdx.x * 16;
  const int lane = threadIdx.x, half = lane >> 4, l15 = lane & 15;
  const int D = nh * dh;
  const float* Qrow = Q  + ((size_t)b * Lq + qi + l15) * D + h * dh;
  const float* Krow = Km + ((size_t)b * Lq + kj + l15) * D + h * dh;

  v8f acc = {};
  for (int k0 = 0; k0 < dh; k0 += 32) {
    v16h af = frag_a16(Qrow + k0, half);
    v16h bf = frag_b16(Krow + k0 + (half ? 16 : 0));   // B = K^T
    acc = __builtin_amdgcn_wmma_f32_16x16x32_f16(
        false, af, false, bf, (short)0, acc, false, false);
  }
  const size_t base = ((size_t)b * nh + h) * Lq * Lq;
  const int col = kj + l15;
  const int r0  = qi + (half ? 8 : 0);
  #pragma unroll
  for (int r = 0; r < 8; ++r) {
    size_t o = base + (size_t)(r0 + r) * Lq + col;
    float v = acc[r] * scale;
    if (bias) v += bias[o];
    S[o] = v;
  }
}

// ---------------------------------------------------------------------------
// A·V:  O[b,qi,h,n] = sum_k S[b,h,qi,k] * V[b,k,h,n]   (K loop over L)
// ---------------------------------------------------------------------------
__global__ __launch_bounds__(32) void attn_av_kernel(
    const float* __restrict__ S, const float* __restrict__ V,
    float* __restrict__ O, int Lq, int nh, int dh)
{
  const int bh = blockIdx.z;
  const int b = bh / nh, h = bh - b * nh;
  const int qi = blockIdx.y * 16, n0 = blockIdx.x * 16;
  const int lane = threadIdx.x, half = lane >> 4, l15 = lane & 15;
  const int D = nh * dh;
  const float* Srow = S + (((size_t)b * nh + h) * Lq + qi + l15) * Lq;
  const float* Vb   = V + (size_t)b * Lq * D + h * dh;

  v8f acc = {};
  for (int k0 = 0; k0 < Lq; k0 += 32) {
    v16h af = frag_a16(Srow + k0, half);
    v16h bf;                                 // V columns are strided by D
    #pragma unroll
    for (int e = 0; e < 16; ++e) {
      int kb = e + (half ? 16 : 0);
      bf[e] = (_Float16)Vb[(size_t)(k0 + kb) * D + n0 + l15];
    }
    acc = __builtin_amdgcn_wmma_f32_16x16x32_f16(
        false, af, false, bf, (short)0, acc, false, false);
  }
  const int col = n0 + l15;
  const int r0  = qi + (half ? 8 : 0);
  #pragma unroll
  for (int r = 0; r < 8; ++r)
    O[(size_t)(b * Lq + r0 + r) * D + h * dh + col] = acc[r];
}

// ---------------------------------------------------------------------------
// Row softmax with key-padding mask. One wave per row of length Ln.
// ---------------------------------------------------------------------------
__global__ __launch_bounds__(32) void softmax_mask_kernel(
    float* __restrict__ S, const unsigned char* __restrict__ mask,
    int Ln, int rows_per_batch, int Lmask)
{
  const int row = blockIdx.x, lane = threadIdx.x;
  const int b = row / rows_per_batch;
  float* Sr = S + (size_t)row * Ln;
  const unsigned char* mb = mask + (size_t)b * Lmask;
  float mx = -3.4e38f;
  for (int c = lane; c < Ln; c += 32) {
    float v = Sr[c];
    if (!mb[c]) v = -1.0e9f;
    Sr[c] = v;
    mx = fmaxf(mx, v);
  }
  mx = wave_max(mx);
  float sum = 0.f;
  for (int c = lane; c < Ln; c += 32) {
    float e = expf(Sr[c] - mx);
    Sr[c] = e; sum += e;
  }
  sum = wave_sum(sum);
  float inv = 1.0f / sum;
  for (int c = lane; c < Ln; c += 32) Sr[c] *= inv;
}

// ---------------------------------------------------------------------------
// LayerNorm with up to two residual inputs: Y = LN(X + R1 + R2)*g + b
// One wave per row; D=256 -> 8 elems per lane.
// ---------------------------------------------------------------------------
__global__ __launch_bounds__(32) void ln_kernel(
    const float* __restrict__ X, const float* __restrict__ R1,
    const float* __restrict__ R2, const float* __restrict__ g,
    const float* __restrict__ bt, float* __restrict__ Y, int D)
{
  const int row = blockIdx.x, lane = threadIdx.x;
  const size_t base = (size_t)row * D;
  float vals[8];
  float s = 0.f;
  #pragma unroll
  for (int i = 0; i < 8; ++i) {
    int c = lane + 32 * i;
    float v = X[base + c];
    if (R1) v += R1[base + c];
    if (R2) v += R2[base + c];
    vals[i] = v; s += v;
  }
  s = wave_sum(s);
  float mean = s / (float)D;
  float vs = 0.f;
  #pragma unroll
  for (int i = 0; i < 8; ++i) { float d = vals[i] - mean; vs += d * d; }
  vs = wave_sum(vs);
  float inv = rsqrtf(vs / (float)D + 1e-5f);
  #pragma unroll
  for (int i = 0; i < 8; ++i) {
    int c = lane + 32 * i;
    Y[base + c] = (vals[i] - mean) * inv * g[c] + bt[c];
  }
}

// Y = (do_gelu ? gelu(X) : X) + (vec ? vec[col] : 0)
__global__ void gelu_addvec_kernel(const float* __restrict__ X,
                                   const float* __restrict__ vec,
                                   float* __restrict__ Y,
                                   int rows, int D, int do_gelu)
{
  size_t i = (size_t)blockIdx.x * blockDim.x + threadIdx.x;
  if (i >= (size_t)rows * D) return;
  float v = X[i];
  if (do_gelu) v = geluf(v);
  if (vec) v += vec[i % D];
  Y[i] = v;
}

// Naive VALU GEMM for skinny matmuls: C = act(A[M,K]W[K,N] + b)
__global__ void naive_gemm_kernel(const float* __restrict__ A,
                                  const float* __restrict__ W,
                                  const float* __restrict__ bias,
                                  float* __restrict__ C,
                                  int M, int N, int K, int act)
{
  int idx = blockIdx.x * blockDim.x + threadIdx.x;
  if (idx >= M * N) return;
  int m = idx / N, n = idx - m * N;
  float s = bias ? bias[n] : 0.f;
  const float* a = A + (size_t)m * K;
  for (int k = 0; k < K; ++k) s += a[k] * W[(size_t)k * N + n];
  C[idx] = apply_act(s, act);
}

// Relative-geometry bias MLP: 3 -> 64 -> NH, written head-major (B,8,L,L)
__global__ __launch_bounds__(256) void rel_bias_kernel(
    const float* __restrict__ raw, const float* __restrict__ w1,
    const float* __restrict__ b1, const float* __restrict__ w2,
    const float* __restrict__ b2, float* __restrict__ bias)
{
  __shared__ float sw1[3 * 64], sb1[64], sw2[64 * 8], sb2[8];
  const int tid = threadIdx.x;
  if (tid < 192) sw1[tid] = w1[tid];
  if (tid < 64)  sb1[tid] = b1[tid];
  sw2[tid] = w2[tid];
  sw2[tid + 256] = w2[tid + 256];
  if (tid < 8) sb2[tid] = b2[tid];
  __syncthreads();

  int idx = blockIdx.x * 256 + tid;
  if (idx >= BB * LL * LL) return;
  int j = idx % LL;
  int t = idx / LL;
  int i = t % LL;
  int b = t / LL;
  float eta_i = raw[(size_t)(b * LL + i) * IN7 + 1];
  float eta_j = raw[(size_t)(b * LL + j) * IN7 + 1];
  float phi_i = raw[(size_t)(b * LL + i) * IN7 + 2];
  float phi_j = raw[(size_t)(b * LL + j) * IN7 + 2];
  float deta = eta_i - eta_j;
  float dphi = atan2f(sinf(phi_i - phi_j), cosf(phi_i - phi_j));
  float dR = sqrtf(deta * deta + dphi * dphi + 1e-8f);
  float acc8[8];
  #pragma unroll
  for (int o = 0; o < 8; ++o) acc8[o] = sb2[o];
  for (int k = 0; k < 64; ++k) {
    float hv = geluf(sw1[k] * deta + sw1[64 + k] * dphi + sw1[128 + k] * dR + sb1[k]);
    #pragma unroll
    for (int o = 0; o < 8; ++o) acc8[o] += hv * sw2[k * 8 + o];
  }
  #pragma unroll
  for (int o = 0; o < 8; ++o)
    bias[(((size_t)b * 8 + o) * LL + i) * LL + j] = acc8[o];
}

// Initial token build from unsmear (bd) / reassign (ba) heads
__global__ void head_init_kernel(const float* __restrict__ hlt,
                                 const float* __restrict__ bd,
                                 const float* __restrict__ ba,
                                 float* __restrict__ tok)
{
  int i = blockIdx.x * blockDim.x + threadIdx.x;
  if (i >= NTOK) return;
  float pt    = fmaxf(hlt[i * 4 + 0], EPSF);
  float eta_h = clampf(hlt[i * 4 + 1], -5.f, 5.f);
  float phi_h = hlt[i * 4 + 2];
  float E     = fmaxf(hlt[i * 4 + 3], EPSF);
  float d_logpt = 0.7f * tanhf(bd[i * 4 + 0]);
  float d_eta   = 0.5f * tanhf(bd[i * 4 + 1]) + 0.3f * tanhf(ba[i * 2 + 0]);
  float d_phi   = 0.5f * tanhf(bd[i * 4 + 2]) + 0.3f * tanhf(ba[i * 2 + 1]);
  float d_logE  = 0.7f * tanhf(bd[i * 4 + 3]);
  float tpt  = expf(clampf(logf(pt) + d_logpt, -9.f, 9.f));
  float teta = clampf(eta_h + d_eta, -5.f, 5.f);
  float tphi = wrap_phi(phi_h + d_phi);
  float tE   = expf(clampf(logf(E) + d_logE, -9.f, 11.f));
  tE = fmaxf(tE, tpt * coshf(teta));
  tok[i * 4 + 0] = tpt; tok[i * 4 + 1] = teta;
  tok[i * 4 + 2] = tphi; tok[i * 4 + 3] = tE;
}

// rep = [log pt, eta, sin phi, cos phi, log E]
__global__ void dn_rep_kernel(const float* __restrict__ tok,
                              float* __restrict__ rep)
{
  int i = blockIdx.x * blockDim.x + threadIdx.x;
  if (i >= NTOK) return;
  float pt  = fmaxf(tok[i * 4 + 0], EPSF);
  float eta = clampf(tok[i * 4 + 1], -5.f, 5.f);
  float phi = tok[i * 4 + 2];
  float E   = fmaxf(tok[i * 4 + 3], EPSF);
  rep[i * 5 + 0] = logf(pt);
  rep[i * 5 + 1] = eta;
  rep[i * 5 + 2] = sinf(phi);
  rep[i * 5 + 3] = cosf(phi);
  rep[i * 5 + 4] = logf(E);
}

// denoiser step update (d already tanh'ed by gemm act)
__global__ void dn_update_kernel(float* __restrict__ tok,
                                 const float* __restrict__ d, float scale)
{
  int i = blockIdx.x * blockDim.x + threadIdx.x;
  if (i >= NTOK) return;
  float pt  = fmaxf(tok[i * 4 + 0], EPSF);
  float eta = clampf(tok[i * 4 + 1], -5.f, 5.f);
  float phi = tok[i * 4 + 2];
  float E   = fmaxf(tok[i * 4 + 3], EPSF);
  float npt  = expf(clampf(logf(pt) + scale * d[i * 4 + 0], -9.f, 9.f));
  float neta = clampf(eta + scale * 0.8f * d[i * 4 + 1], -5.f, 5.f);
  float nphi = wrap_phi(phi + scale * 0.8f * d[i * 4 + 2]);
  float nE   = expf(clampf(logf(E) + scale * d[i * 4 + 3], -9.f, 11.f));
  nE = fmaxf(nE, npt * coshf(neta));
  tok[i * 4 + 0] = npt; tok[i * 4 + 1] = neta;
  tok[i * 4 + 2] = nphi; tok[i * 4 + 3] = nE;
}

__global__ void copy_kernel(const float* __restrict__ src,
                            float* __restrict__ dst, int n)
{
  int i = blockIdx.x * blockDim.x + threadIdx.x;
  if (i < n) dst[i] = src[i];
}

// tok_w = sigmoid(x . action_w + b) * mask
__global__ void tokw_kernel(const float* __restrict__ x,
                            const float* __restrict__ aw,
                            const float* __restrict__ ab,
                            const unsigned char* __restrict__ mask,
                            float* __restrict__ out)
{
  int i = blockIdx.x * blockDim.x + threadIdx.x;
  if (i >= NTOK) return;
  float s = ab[0];
  const float* xr = x + (size_t)i * DD;
  for (int k = 0; k < DD; ++k) s += xr[k] * aw[k];
  float sig = 1.0f / (1.0f + expf(-s));
  out[i] = sig * (mask[i] ? 1.0f : 0.0f);
}

// q_vec = pool_query (1x256) @ wq + bq    (batch-independent)
__global__ void pool_q_kernel(const float* __restrict__ pq,
                              const float* __restrict__ wq,
                              const float* __restrict__ bq,
                              float* __restrict__ qv)
{
  int n = threadIdx.x;
  float s = bq[n];
  for (int k = 0; k < DD; ++k) s += pq[k] * wq[(size_t)k * DD + n];
  qv[n] = s;
}

// pool scores: ps[b,h,t] = qv[h,:] . K[b,t,h,:] / 8   (4 heads, dh=64)
__global__ void pool_scores_kernel(const float* __restrict__ qv,
                                   const float* __restrict__ Kp,
                                   float* __restrict__ ps)
{
  int idx = blockIdx.x * blockDim.x + threadIdx.x;
  if (idx >= BB * 4 * LL) return;
  int t = idx % LL;
  int r = idx / LL;
  int h = r % 4;
  int b = r / 4;
  const float* kp = Kp + (size_t)(b * LL + t) * DD + h * 64;
  const float* qh = qv + h * 64;
  float s = 0.f;
  for (int d = 0; d < 64; ++d) s += qh[d] * kp[d];
  ps[idx] = s * 0.125f;
}

// po[b, h*64+d] = sum_k ps[b,h,k] * V[b,k,h,d]
__global__ void pool_out_kernel(const float* __restrict__ ps,
                                const float* __restrict__ Vp,
                                float* __restrict__ po)
{
  int idx = blockIdx.x * blockDim.x + threadIdx.x;
  if (idx >= BB * DD) return;
  int c = idx % DD;
  int b = idx / DD;
  int h = c >> 6;
  float s = 0.f;
  const float* pr = ps + (size_t)(b * 4 + h) * LL;
  for (int k = 0; k < LL; ++k)
    s += pr[k] * Vp[(size_t)(b * LL + k) * DD + c];
  po[idx] = s;
}

// ---------------------------------------------------------------------------
// Host orchestration
// ---------------------------------------------------------------------------
extern "C" void kernel_launch(void* const* d_in, const int* in_sizes, int n_in,
                              void* d_out, int out_size, void* d_ws, size_t ws_size,
                              hipStream_t stream)
{
  if (n_in < 182 || !d_out || !d_ws) return;
  auto F = [&](int i) -> const float* { return (const float*)d_in[i]; };

  // param leaf indices (dict insertion order)
  enum {
    I_RAW = 0, I_HLT = 1, I_MASK = 2,
    I_IN_W = 3, I_IN_B = 4, I_INLN_G = 5, I_INLN_B = 6,
    I_REL1_W = 7, I_REL1_B = 8, I_REL2_W = 9, I_REL2_B = 10,
    I_LAYER0 = 11,                               // 16 leaves x 8 layers
    I_TOKNORM_G = 139, I_TOKNORM_B = 140,
    I_ACT_W = 141, I_ACT_B = 142,
    I_UN1_W = 143, I_UN1_B = 144, I_UN2_W = 145, I_UN2_B = 146,
    I_RE1_W = 147, I_RE1_B = 148, I_RE2_W = 149, I_RE2_B = 150,
    I_HYP = 151, I_POOLQ = 152,
    I_PA_WQ = 153, I_PA_BQ = 154, I_PA_WK = 155, I_PA_BK = 156,
    I_PA_WV = 157, I_PA_BV = 158, I_PA_WO = 159, I_PA_BO = 160,
    I_DN_STEP = 161,
    I_DNIN_W = 162, I_DNIN_B = 163, I_DNLN_G = 164, I_DNLN_B = 165,
    I_DA_WQ = 166, I_DA_BQ = 167, I_DA_WK = 168, I_DA_BK = 169,
    I_DA_WV = 170, I_DA_BV = 171, I_DA_WO = 172, I_DA_BO = 173,
    I_DNF1_W = 174, I_DNF1_B = 175, I_DNF2_W = 176, I_DNF2_B = 177,
    I_DNN_G = 178, I_DNN_B = 179, I_DNH_W = 180, I_DNH_B = 181
  };

  const float* const_raw = F(I_RAW);
  const float* const_hlt = F(I_HLT);
  const unsigned char* mask = (const unsigned char*)d_in[I_MASK];
  float* out = (float*)d_out;

  // ---- workspace carve (floats) ----
  float* ws = (float*)d_ws;
  size_t off = 0;
  auto alloc = [&](size_t n) -> float* { float* p = ws + off; off += n; return p; };
  const size_t ND = (size_t)NTOK * DD;
  float* xb0 = alloc(ND);
  float* xb1 = alloc(ND);
  float* xb2 = alloc(ND);
  float* xb3 = alloc(ND);
  float* qb  = alloc(ND);
  float* kb  = alloc(ND);
  float* vb  = alloc(ND);
  float* ffb = alloc((size_t)NTOK * FF_);
  float* biasb   = alloc((size_t)BB * NH8 * LL * LL);
  float* scoresb = alloc((size_t)BB * NH8 * LL * LL);
  float* xe  = alloc(ND);
  float* kdn = alloc(ND);
  float* vdn = alloc(ND);
  float* tokb = alloc((size_t)NTOK * 4);
  float* repb = alloc((size_t)NTOK * 5);
  float* bdb  = alloc((size_t)NTOK * 4);
  float* bab  = alloc((size_t)NTOK * 2);
  float* qvec = alloc(256);
  float* psb  = alloc((size_t)BB * 4 * LL);
  float* pob  = alloc((size_t)BB * DD);
  if (off * sizeof(float) > ws_size) return;

  auto wgemm = [&](const float* A, const float* W, const float* bias, float* C,
                   int M, int N, int K, int act) {
    dim3 g(N / 32, M / 64);
    wmma_gemm_kernel<<<g, 128, 0, stream>>>(A, W, bias, C, M, N, K, act);
  };
  auto ngemm = [&](const float* A, const float* W, const float* bias, float* C,
                   int M, int N, int K, int act) {
    naive_gemm_kernel<<<(M * N + 255) / 256, 256, 0, stream>>>(A, W, bias, C, M, N, K, act);
  };
  auto ln = [&](const float* X, const float* R1, const float* R2,
                int gi, int bi, float* Y) {
    ln_kernel<<<NTOK, 32, 0, stream>>>(X, R1, R2, F(gi), F(bi), Y, DD);
  };
  auto elts = [&](size_t n) { return (int)((n + 255) / 256); };

  // ---- input embed: x = gelu(ln(lin(raw))) ----
  ngemm(const_raw, F(I_IN_W), F(I_IN_B), xb0, NTOK, DD, IN7, 0);
  ln(xb0, nullptr, nullptr, I_INLN_G, I_INLN_B, xb1);
  gelu_addvec_kernel<<<elts(ND), 256, 0, stream>>>(xb1, nullptr, xb1, NTOK, DD, 1);

  // ---- relative-geometry attention bias (B,8,L,L) ----
  rel_bias_kernel<<<elts((size_t)BB * LL * LL), 256, 0, stream>>>(
      const_raw, F(I_REL1_W), F(I_REL1_B), F(I_REL2_W), F(I_REL2_B), biasb);

  // ---- encoder: 8 transformer layers (WMMA) ----
  float* x = xb1;
  for (int l = 0; l < 8; ++l) {
    const int p = I_LAYER0 + l * 16;  // wq bq wk bk wv bv wo bo ln1g ln1b ff1w ff1b ff2w ff2b ln2g ln2b
    wgemm(x, F(p + 0), F(p + 1), qb, NTOK, DD, DD, 0);
    wgemm(x, F(p + 2), F(p + 3), kb, NTOK, DD, DD, 0);
    wgemm(x, F(p + 4), F(p + 5), vb, NTOK, DD, DD, 0);
    {
      dim3 g(LL / 16, LL / 16, BB * NH8);
      attn_scores_kernel<<<g, 32, 0, stream>>>(qb, kb, biasb, scoresb,
                                               LL, NH8, 32, 0.1767766953f);
    }
    softmax_mask_kernel<<<BB * NH8 * LL, 32, 0, stream>>>(scoresb, mask, LL, NH8 * LL, LL);
    {
      dim3 g(32 / 16, LL / 16, BB * NH8);
      attn_av_kernel<<<g, 32, 0, stream>>>(scoresb, vb, xb2, LL, NH8, 32);
    }
    wgemm(xb2, F(p + 6), F(p + 7), xb3, NTOK, DD, DD, 0);
    ln(x, xb3, nullptr, p + 8, p + 9, xb0);                  // x2 = LN(x + attn)
    wgemm(xb0, F(p + 10), F(p + 11), ffb, NTOK, FF_, DD, 1); // gelu(ff1)
    wgemm(ffb, F(p + 12), F(p + 13), xb3, NTOK, DD, FF_, 0);
    ln(xb0, xb3, nullptr, p + 14, p + 15, xb1);              // x = LN(x2 + ff)
    x = xb1;
  }

  // ---- token norm + hypothesis embed ----
  ln(x, nullptr, nullptr, I_TOKNORM_G, I_TOKNORM_B, xb0);
  gelu_addvec_kernel<<<elts(ND), 256, 0, stream>>>(xb0, F(I_HYP), xe, NTOK, DD, 0);

  // ---- unsmear / reassign heads -> initial tok ----
  wgemm(xe, F(I_UN1_W), F(I_UN1_B), xb2, NTOK, DD, DD, 1);
  ngemm(xb2, F(I_UN2_W), F(I_UN2_B), bdb, NTOK, 4, DD, 0);
  wgemm(xe, F(I_RE1_W), F(I_RE1_B), xb3, NTOK, 128, DD, 1);
  ngemm(xb3, F(I_RE2_W), F(I_RE2_B), bab, NTOK, 2, 128, 0);
  head_init_kernel<<<elts(NTOK), 256, 0, stream>>>(const_hlt, bdb, bab, tokb);

  // ---- denoiser (3 steps, 4-head cross-attn, dh=64). K/V of xe are static ----
  wgemm(xe, F(I_DA_WK), F(I_DA_BK), kdn, NTOK, DD, DD, 0);
  wgemm(xe, F(I_DA_WV), F(I_DA_BV), vdn, NTOK, DD, DD, 0);
  const float dscale = 0.35f / 3.0f;
  for (int s = 0; s < 3; ++s) {
    dn_rep_kernel<<<elts(NTOK), 256, 0, stream>>>(tokb, repb);
    ngemm(repb, F(I_DNIN_W), F(I_DNIN_B), xb0, NTOK, DD, 5, 0);
    ln(xb0, nullptr, nullptr, I_DNLN_G, I_DNLN_B, xb1);
    gelu_addvec_kernel<<<elts(ND), 256, 0, stream>>>(
        xb1, F(I_DN_STEP) + (size_t)s * DD, xb1, NTOK, DD, 1);   // q
    wgemm(xb1, F(I_DA_WQ), F(I_DA_BQ), qb, NTOK, DD, DD, 0);
    {
      dim3 g(LL / 16, LL / 16, BB * 4);
      attn_scores_kernel<<<g, 32, 0, stream>>>(qb, kdn, nullptr, scoresb,
                                               LL, 4, 64, 0.125f);
    }
    softmax_mask_kernel<<<BB * 4 * LL, 32, 0, stream>>>(scoresb, mask, LL, 4 * LL, LL);
    {
      dim3 g(64 / 16, LL / 16, BB * 4);
      attn_av_kernel<<<g, 32, 0, stream>>>(scoresb, vdn, xb2, LL, 4, 64);
    }
    wgemm(xb2, F(I_DA_WO), F(I_DA_BO), xb3, NTOK, DD, DD, 0);   // ctx
    wgemm(xb3, F(I_DNF1_W), F(I_DNF1_B), xb0, NTOK, DD, DD, 1); // gelu(ff1(ctx))
    wgemm(xb0, F(I_DNF2_W), F(I_DNF2_B), ffb, NTOK, DD, DD, 0);
    ln(xb1, xb3, ffb, I_DNN_G, I_DNN_B, xb0);                   // h = LN(q+ctx+ff)
    ngemm(xb0, F(I_DNH_W), F(I_DNH_B), bdb, NTOK, 4, DD, 2);    // tanh head
    dn_update_kernel<<<elts(NTOK), 256, 0, stream>>>(tokb, bdb, dscale);
  }
  copy_kernel<<<elts((size_t)NTOK * 4), 256, 0, stream>>>(tokb, out, NTOK * 4);

  // ---- token weights ----
  tokw_kernel<<<elts(NTOK), 256, 0, stream>>>(xe, F(I_ACT_W), F(I_ACT_B),
                                              mask, out + (size_t)NTOK * 4);

  // ---- attention pooling (4 heads, dh=64, single query) ----
  wgemm(xe, F(I_PA_WK), F(I_PA_BK), kb, NTOK, DD, DD, 0);
  wgemm(xe, F(I_PA_WV), F(I_PA_BV), vb, NTOK, DD, DD, 0);
  pool_q_kernel<<<1, 256, 0, stream>>>(F(I_POOLQ), F(I_PA_WQ), F(I_PA_BQ), qvec);
  pool_scores_kernel<<<elts((size_t)BB * 4 * LL), 256, 0, stream>>>(qvec, kb, psb);
  softmax_mask_kernel<<<BB * 4, 32, 0, stream>>>(psb, mask, LL, 4, LL);
  pool_out_kernel<<<elts((size_t)BB * DD), 256, 0, stream>>>(psb, vb, pob);
  ngemm(pob, F(I_PA_WO), F(I_PA_BO), out + (size_t)NTOK * 4 + NTOK, BB, DD, DD, 0);
}
